// TrajectoryFlowMap_78288663872186
// MI455X (gfx1250) — matrix operand order repaired and checked by reading, not compiled, not run on hardware
//
#include <hip/hip_runtime.h>

#define NN 16384
#define KC 128          // K-chunk (columns of T per iteration)
#define PSTR 136        // padded LDS row stride in halfs (272B: 16B-aligned, odd-dword stride/4 -> few bank conflicts)

typedef __attribute__((ext_vector_type(16))) _Float16 v16h;
typedef __attribute__((ext_vector_type(8)))  float    v8f;
typedef __attribute__((ext_vector_type(4)))  float    f32x4;   // native vector for nontemporal builtins

union Frag  { unsigned u[8];  v16h h;  };
union PPack { _Float16 h[16]; uint4 q[2]; };

// ---------------- Kernel 1: Gt[d][j] = (f16) emb[x[j]][d]  (gather + transpose + cvt) ----------------
__global__ __launch_bounds__(256) void gather_tr_kernel(const int* __restrict__ x,
                                                        const float* __restrict__ emb,
                                                        _Float16* __restrict__ Gt) {
  int j = blockIdx.x * 256 + threadIdx.x;
  int src = x[j];
  const f32x4* row = (const f32x4*)(emb + (size_t)src * 64);
#pragma unroll
  for (int q = 0; q < 16; ++q) {
    f32x4 v = row[q];
    Gt[(size_t)(4 * q + 0) * NN + j] = (_Float16)v.x;   // writes coalesced over j
    Gt[(size_t)(4 * q + 1) * NN + j] = (_Float16)v.y;
    Gt[(size_t)(4 * q + 2) * NN + j] = (_Float16)v.z;
    Gt[(size_t)(4 * q + 3) * NN + j] = (_Float16)v.w;
  }
}

// ---------------- Kernel 2: fused online-softmax x WMMA matmul ----------------
// Block = 256 threads (8 waves), owns 32 output rows.
// wave w: output tile rows [rowbase, rowbase+16), cols [d0, d0+16)
__global__ __launch_bounds__(256) void flow_kernel(const float* __restrict__ T,
                                                   const _Float16* __restrict__ Gt,
                                                   float* __restrict__ out) {
  __shared__ _Float16 Pl[32 * PSTR];   // probabilities (f16), padded rows
  __shared__ _Float16 Gl[64 * PSTR];   // gathered-embedding chunk, transposed: [d][jlocal]
  __shared__ float m_s[32], l_s[32], sc_s[32];

  const int tid  = threadIdx.x;
  const int lane = tid & 31;
  const int wave = tid >> 5;
  const int r    = tid >> 3;           // softmax row owned by this thread (0..31)
  const int sub  = tid & 7;            // 8 threads per row
  const int rowbase = (wave >> 2) << 4;
  const int d0      = (wave & 3) << 4;
  const int r0      = blockIdx.x * 32;

  if (tid < 32) { m_s[tid] = -__builtin_inff(); l_s[tid] = 0.0f; }

  v8f c = {};  // fp32 accumulator tile (16x16)

  for (int k0 = 0; k0 < NN; k0 += KC) {
    __syncthreads();   // previous chunk's LDS reads done; also covers init on first iter

    // ---- stage Gt chunk [64 d][KC j] into LDS (coalesced 16B loads/stores) ----
    {
      int d = tid >> 2, q = tid & 3;
      const uint4* src = (const uint4*)(Gt + (size_t)d * NN + k0 + q * 32);
      uint4* dst = (uint4*)(Gl + d * PSTR + q * 32);
#pragma unroll
      for (int i = 0; i < 4; ++i) dst[i] = src[i];
    }

    // ---- online softmax over this chunk: each thread handles 16 cols of one row ----
    const f32x4* trow = (const f32x4*)(T + (size_t)(r0 + r) * NN + k0 + sub * 16);
    float v[16];
#pragma unroll
    for (int q = 0; q < 4; ++q) {
      f32x4 t4 = __builtin_nontemporal_load(trow + q);   // NT: keep Gt resident in L2
      v[4*q+0] = t4.x; v[4*q+1] = t4.y; v[4*q+2] = t4.z; v[4*q+3] = t4.w;
    }
    float bm = v[0];
#pragma unroll
    for (int i = 1; i < 16; ++i) bm = fmaxf(bm, v[i]);
    bm = fmaxf(bm, __shfl_xor(bm, 1));
    bm = fmaxf(bm, __shfl_xor(bm, 2));
    bm = fmaxf(bm, __shfl_xor(bm, 4));

    float m_old = m_s[r];
    float m_new = fmaxf(m_old, bm);
    float scale = __expf(m_old - m_new);   // first chunk: exp(-inf)=0

    PPack pk;
    float bs = 0.0f;
#pragma unroll
    for (int i = 0; i < 16; ++i) {
      float e = __expf(v[i] - m_new);
      bs += e;
      pk.h[i] = (_Float16)e;
    }
    bs += __shfl_xor(bs, 1);
    bs += __shfl_xor(bs, 2);
    bs += __shfl_xor(bs, 4);

    if (sub == 0) { m_s[r] = m_new; sc_s[r] = scale; l_s[r] = l_s[r] * scale + bs; }

    // store probabilities (16 halfs = two b128 stores, 16B-aligned)
    uint4* pd = (uint4*)(Pl + r * PSTR + sub * 16);
    pd[0] = pk.q[0];
    pd[1] = pk.q[1];

    __syncthreads();

    // ---- WMMA phase: rescale accumulators, then 4 K-steps of 16x16x32 ----
#pragma unroll
    for (int g = 0; g < 8; ++g) {
      int mr = rowbase + g + ((lane & 16) ? 8 : 0);   // C/D layout: M = g + (lane>=16 ? 8 : 0)
      c[g] *= sc_s[mr];
    }

#pragma unroll
    for (int kk = 0; kk < KC; kk += 32) {
      Frag a, b;
      // A fragment (16x32 f16): lane m = rowbase + lane%16;
      // v0-7: K = (lane>=16?16:0)+2v (contiguous, matches B)
      {
        int m  = rowbase + (lane & 15);
        int ak = kk + ((lane & 16) ? 16 : 0);
        const _Float16* base = Pl + m * PSTR + ak;
#pragma unroll
        for (int vv = 0; vv < 4; ++vv) {
          a.u[vv]     = *(const unsigned*)(base + 2 * vv);
          a.u[vv + 4] = *(const unsigned*)(base + 8 + 2 * vv);
        }
      }
      // B fragment (32x16 f16): lane col d = d0 + lane%16;
      // lanes 0-15: K=0..15, lanes 16-31: K=16..31, 2 K per VGPR
      {
        int d  = d0 + (lane & 15);
        int bk = kk + ((lane & 16) ? 16 : 0);
        const _Float16* base = Gl + d * PSTR + bk;
#pragma unroll
        for (int vv = 0; vv < 8; ++vv)
          b.u[vv] = *(const unsigned*)(base + 2 * vv);
      }
      c = __builtin_amdgcn_wmma_f32_16x16x32_f16(false, a.h, false, b.h,
                                                 (short)0, c, false, false);
    }
  }

  // ---- epilogue: divide by running sum, store ----
#pragma unroll
  for (int g = 0; g < 8; ++g) {
    int mr = rowbase + g + ((lane & 16) ? 8 : 0);
    float o = c[g] / l_s[mr];
    __builtin_nontemporal_store(o, out + (size_t)(r0 + mr) * 64 + d0 + (lane & 15));
  }
}

extern "C" void kernel_launch(void* const* d_in, const int* in_sizes, int n_in,
                              void* d_out, int out_size, void* d_ws, size_t ws_size,
                              hipStream_t stream) {
  const int*   x   = (const int*)d_in[0];
  const float* emb = (const float*)d_in[1];
  const float* T   = (const float*)d_in[2];
  float* out = (float*)d_out;
  _Float16* Gt = (_Float16*)d_ws;   // 64 x 16384 f16 = 2 MB scratch

  gather_tr_kernel<<<NN / 256, 256, 0, stream>>>(x, emb, Gt);
  flow_kernel<<<NN / 32, 256, 0, stream>>>(T, Gt, out);
}
